// LSTMCell_19009525252387
// MI455X (gfx1250) — matrix operand order repaired and checked by reading, not compiled, not run on hardware
//
#include <hip/hip_runtime.h>

#define B_SZ   8192
#define IN_SZ  2048
#define H_SZ   2048
#define K_SZ   4096      // IN + H
#define MT     128       // block tile M
#define NT     64        // block tile N (per gate)
#define KT     32        // K step (one bf16 WMMA K)
#define LDSK   40        // padded LDS row stride in bf16 elems (80B -> conflict-free)
#define NKT    (K_SZ / KT)

#define ABUF_E (MT * LDSK)          // elems per A buffer
#define BBUF_E (4 * NT * LDSK)      // elems per B buffer
#define ABUF_BYTES (ABUF_E * 2)
#define BBUF_BYTES (BBUF_E * 2)

typedef __attribute__((ext_vector_type(8)))  float   v8f;
typedef __attribute__((ext_vector_type(8)))  __bf16  bf16x8;
typedef __attribute__((ext_vector_type(16))) __bf16  bf16x16;

__device__ __forceinline__ unsigned short f2bf(float f) {
  unsigned int u = __float_as_uint(f);
  u += 0x7fffu + ((u >> 16) & 1u);      // round-to-nearest-even
  return (unsigned short)(u >> 16);
}

// low 32 bits of a generic LDS pointer == LDS byte address for DS/async ops
__device__ __forceinline__ unsigned lds_addr32(const void* p) {
  return (unsigned)(unsigned long long)p;
}

// async DMA: global (saddr base + 32-bit voffset) -> LDS (32-bit lds addr), 16B/lane
__device__ __forceinline__ void async_ld_b128(unsigned lds, unsigned goff, const void* base) {
  asm volatile("global_load_async_to_lds_b128 %0, %1, %2"
               :
               : "v"(lds), "v"(goff), "s"(base)
               : "memory");
}

__device__ __forceinline__ void wait_async0() {
  asm volatile("s_wait_asynccnt 0x0" ::: "memory");
}

// ---------------- pack kernels ----------------

__global__ void pack_combined_kernel(const float* __restrict__ x,
                                     const float* __restrict__ h,
                                     unsigned short* __restrict__ cmb) {
  const size_t total = (size_t)B_SZ * K_SZ / 4;
  const size_t stride = (size_t)gridDim.x * blockDim.x;
  for (size_t i = (size_t)blockIdx.x * blockDim.x + threadIdx.x; i < total; i += stride) {
    const size_t e   = i * 4;
    const size_t b   = e / K_SZ;
    const size_t col = e % K_SZ;
    const float* src = (col < IN_SZ) ? (x + b * IN_SZ + col)
                                     : (h + b * H_SZ + (col - IN_SZ));
    const float4 v = *(const float4*)src;
    ushort4 o;
    o.x = f2bf(v.x); o.y = f2bf(v.y); o.z = f2bf(v.z); o.w = f2bf(v.w);
    *(ushort4*)(cmb + e) = o;
  }
}

__global__ void pack_weights_kernel(const float* __restrict__ w0,  // W_i
                                    const float* __restrict__ w1,  // W_f
                                    const float* __restrict__ w2,  // W_o
                                    const float* __restrict__ w3,  // W_c
                                    unsigned short* __restrict__ wb) {
  const size_t per_gate = (size_t)H_SZ * K_SZ;
  const size_t total = 4 * per_gate / 4;
  const size_t stride = (size_t)gridDim.x * blockDim.x;
  for (size_t i = (size_t)blockIdx.x * blockDim.x + threadIdx.x; i < total; i += stride) {
    const size_t e = i * 4;
    const size_t g = e / per_gate;
    const size_t r = e % per_gate;
    const float* w = (g == 0) ? w0 : (g == 1) ? w1 : (g == 2) ? w2 : w3;
    const float4 v = *(const float4*)(w + r);
    ushort4 o;
    o.x = f2bf(v.x); o.y = f2bf(v.y); o.z = f2bf(v.z); o.w = f2bf(v.w);
    *(ushort4*)(wb + e) = o;
  }
}

__global__ void pack_bias_kernel(const float* __restrict__ b0,  // b_i
                                 const float* __restrict__ b1,  // b_f
                                 const float* __restrict__ b2,  // b_o
                                 const float* __restrict__ b3,  // b_c
                                 float* __restrict__ out) {
  const int i = blockIdx.x * blockDim.x + threadIdx.x;
  if (i < 4 * H_SZ) {
    const int g = i / H_SZ;
    const int r = i % H_SZ;
    const float* b = (g == 0) ? b0 : (g == 1) ? b1 : (g == 2) ? b2 : b3;
    out[i] = b[r];
  }
}

// ---------------- fused bf16 WMMA GEMM + LSTM epilogue ----------------

__global__ __launch_bounds__(256) void lstm_wmma_kernel(
    const unsigned short* __restrict__ cmb,    // [B][K] bf16
    const unsigned short* __restrict__ wgt,    // [4][H][K] bf16 (i,f,o,c)
    const float* __restrict__ biasc,           // [4][H]
    const float* __restrict__ c_old,           // [B][H]
    float* __restrict__ h_new,                 // [B][H]
    float* __restrict__ c_new)                 // [B][H]
{
  __shared__ __attribute__((aligned(16))) unsigned short sA[2 * ABUF_E];
  __shared__ __attribute__((aligned(16))) unsigned short sB[2 * BBUF_E];

  const int tid  = threadIdx.x;
  const int lane = tid & 31;
  const int wv   = tid >> 5;     // wave id 0..7
  const int wm   = wv & 3;       // wave M coordinate (0..3)
  const int wn   = wv >> 2;      // wave N coordinate (0..1)
  const int l16  = lane & 15;
  const int kg   = lane >> 4;    // lane half (K-group)

  const int m0 = blockIdx.x * MT;
  const int n0 = blockIdx.y * NT;

  // ---- per-thread DMA chunk mapping: 16B chunks (8 bf16) ----
  unsigned aoff[2], alds[2];           // global byte offset / LDS byte addr (buf0)
#pragma unroll
  for (int i = 0; i < 2; ++i) {
    const int cidx = tid + 256 * i;    // 0..511 = 128 rows * 4 chunks
    const int row  = cidx >> 2;
    const int kc   = cidx & 3;
    aoff[i] = (unsigned)(((size_t)(m0 + row) * K_SZ + kc * 8) * 2);
    alds[i] = lds_addr32(sA + row * LDSK + kc * 8);
  }
  unsigned boff[4], blds[4];
#pragma unroll
  for (int i = 0; i < 4; ++i) {
    const int cidx = tid + 256 * i;    // 0..1023 = 256 rows * 4 chunks
    const int rt   = cidx >> 2;        // 0..255
    const int kc   = cidx & 3;
    const int g    = rt >> 6;
    const int row  = rt & 63;
    boff[i] = (unsigned)((((size_t)g * H_SZ + (size_t)(n0 + row)) * K_SZ + kc * 8) * 2);
    blds[i] = lds_addr32(sB + (g * NT + row) * LDSK + kc * 8);
  }

  // issue one K-tile's async DMA into buffer `buf`
  auto issue_tile = [&](int kt, int buf) {
    const unsigned kb = (unsigned)kt * (KT * 2);         // byte advance along K
    const unsigned ab = (unsigned)buf * ABUF_BYTES;
    const unsigned bb = (unsigned)buf * BBUF_BYTES;
#pragma unroll
    for (int i = 0; i < 2; ++i) async_ld_b128(alds[i] + ab, aoff[i] + kb, cmb);
#pragma unroll
    for (int i = 0; i < 4; ++i) async_ld_b128(blds[i] + bb, boff[i] + kb, wgt);
  };

  v8f acc[4][2][2];
#pragma unroll
  for (int g = 0; g < 4; ++g)
#pragma unroll
    for (int mi = 0; mi < 2; ++mi)
#pragma unroll
      for (int ni = 0; ni < 2; ++ni)
#pragma unroll
        for (int e = 0; e < 8; ++e) acc[g][mi][ni][e] = 0.0f;

  issue_tile(0, 0);                       // prime the pipeline

  for (int kt = 0; kt < NKT; ++kt) {
    const int buf = kt & 1;
    wait_async0();                        // this wave's DMA for tile kt is in LDS
    __syncthreads();                      // all waves: DMA done & prev compute done
    if (kt + 1 < NKT) issue_tile(kt + 1, buf ^ 1);   // stream next tile into other buffer

    const unsigned short* bufA = sA + buf * ABUF_E;
    const unsigned short* bufB = sB + buf * BBUF_E;

    // A fragments: lane half kg holds K = kg*8 + {0..7} and 16 + kg*8 + {0..7}
    bf16x16 afr[2];
#pragma unroll
    for (int mi = 0; mi < 2; ++mi) {
      const unsigned short* p = bufA + (wm * 32 + mi * 16 + l16) * LDSK;
      const bf16x8 lo = *(const bf16x8*)(p + kg * 8);
      const bf16x8 hi = *(const bf16x8*)(p + 16 + kg * 8);
#pragma unroll
      for (int e = 0; e < 8; ++e) { afr[mi][e] = lo[e]; afr[mi][e + 8] = hi[e]; }
    }

    // B fragments: lane half kg holds K = kg*8 + {0..7} and 16 + kg*8 + {0..7} at column l16
#pragma unroll
    for (int g = 0; g < 4; ++g) {
#pragma unroll
      for (int ni = 0; ni < 2; ++ni) {
        const unsigned short* p =
            bufB + (g * NT + wn * 32 + ni * 16 + l16) * LDSK;
        const bf16x8 lo = *(const bf16x8*)(p + kg * 8);
        const bf16x8 hi = *(const bf16x8*)(p + 16 + kg * 8);
        bf16x16 bfr;
#pragma unroll
        for (int e = 0; e < 8; ++e) { bfr[e] = lo[e]; bfr[e + 8] = hi[e]; }
#pragma unroll
        for (int mi = 0; mi < 2; ++mi) {
          acc[g][mi][ni] = __builtin_amdgcn_wmma_f32_16x16x32_bf16(
              false, afr[mi], false, bfr, (short)0, acc[g][mi][ni], false, false);
        }
      }
    }
  }

  // ---- fused LSTM epilogue ----
  const int mbase = m0 + wm * 32;
  const int jwave = n0 + wn * 32;
#pragma unroll
  for (int mi = 0; mi < 2; ++mi) {
#pragma unroll
    for (int ni = 0; ni < 2; ++ni) {
      const int j   = jwave + ni * 16 + l16;
      const float bi = biasc[0 * H_SZ + j];
      const float bf = biasc[1 * H_SZ + j];
      const float bo = biasc[2 * H_SZ + j];
      const float bc = biasc[3 * H_SZ + j];
#pragma unroll
      for (int e = 0; e < 8; ++e) {
        const int m = mbase + mi * 16 + kg * 8 + e;   // C/D layout: lanes>=16 -> M+8
        const size_t ix = (size_t)m * H_SZ + j;
        const float gi = acc[0][mi][ni][e] + bi;
        const float gf = acc[1][mi][ni][e] + bf;
        const float go = acc[2][mi][ni][e] + bo;
        const float gc = acc[3][mi][ni][e] + bc;
        const float si = 1.0f / (1.0f + __expf(-gi));
        const float sf = 1.0f / (1.0f + __expf(-gf));
        const float so = 1.0f / (1.0f + __expf(-go));
        const float e2 = __expf(2.0f * gc);
        const float ch = (e2 - 1.0f) / (e2 + 1.0f);   // tanh(gc)
        const float cn = sf * c_old[ix] + si * ch;
        const float e3 = __expf(2.0f * cn);
        const float th = (e3 - 1.0f) / (e3 + 1.0f);   // tanh(c_new)
        c_new[ix] = cn;
        h_new[ix] = so * th;
      }
    }
  }
}

// ---------------- launch ----------------

extern "C" void kernel_launch(void* const* d_in, const int* in_sizes, int n_in,
                              void* d_out, int out_size, void* d_ws, size_t ws_size,
                              hipStream_t stream) {
  const float* x   = (const float*)d_in[0];
  const float* h   = (const float*)d_in[1];
  const float* c   = (const float*)d_in[2];
  const float* Wi  = (const float*)d_in[3];
  const float* bi  = (const float*)d_in[4];
  const float* Wf  = (const float*)d_in[5];
  const float* bf  = (const float*)d_in[6];
  const float* Wc  = (const float*)d_in[7];
  const float* bc  = (const float*)d_in[8];
  const float* Wo  = (const float*)d_in[9];
  const float* bo  = (const float*)d_in[10];

  // workspace layout: cmb bf16 [B][K] | W bf16 [4][H][K] | bias f32 [4][H]
  unsigned short* cmb = (unsigned short*)d_ws;
  unsigned short* wb  = cmb + (size_t)B_SZ * K_SZ;
  float* biasc        = (float*)(wb + (size_t)4 * H_SZ * K_SZ);

  float* hn = (float*)d_out;                       // h_new first (tuple order)
  float* cn = hn + (size_t)B_SZ * H_SZ;            // then c_new

  pack_combined_kernel<<<4096, 256, 0, stream>>>(x, h, cmb);
  pack_weights_kernel<<<4096, 256, 0, stream>>>(Wi, Wf, Wo, Wc, wb);  // gate order i,f,o,c
  pack_bias_kernel<<<32, 256, 0, stream>>>(bi, bf, bo, bc, biasc);

  dim3 grid(B_SZ / MT, H_SZ / NT);
  lstm_wmma_kernel<<<grid, 256, 0, stream>>>(cmb, wb, biasc, c, hn, cn);
}
